// GraphTrajectoryLSTM_73967926772364
// MI455X (gfx1250) — matrix operand order repaired and checked by reading, not compiled, not run on hardware
//
#include <hip/hip_runtime.h>
#include <math.h>

// ---------------------------------------------------------------------------
// GraphTrajectoryLSTM for MI455X (gfx1250, wave32, WMMA f16->f32 accumulate)
// ---------------------------------------------------------------------------

typedef __attribute__((ext_vector_type(16))) _Float16 v16h;
typedef __attribute__((ext_vector_type(8)))  float    v8f;

#define B_   256
#define N_   200
#define E_   1600
#define H_   256
#define T_   30
#define L_   30
#define G4_  1024          // 4*H gates
#define BN_  (B_ * N_)     // 51200

__device__ __forceinline__ float sigmoidf_(float x) { return 1.0f / (1.0f + __expf(-x)); }

__device__ __forceinline__ v8f wmma32(v16h a, v16h b, v8f c) {
  // D = A(16x32 f16) * B(32x16 f16) + C(f32)
  return __builtin_amdgcn_wmma_f32_16x16x32_f16(false, a, false, b, (short)0, c, false, false);
}

// A fragment (16x32, row-major source, lda in elements).
// lane m = lane%16; elements 0..15 -> K = k0 + 16*(lane>=16) + e.
__device__ __forceinline__ v16h load_a_frag(const _Float16* __restrict__ A, long lda,
                                            int m0, int k0) {
  int lane = threadIdx.x & 31;
  const _Float16* p = A + (long)(m0 + (lane & 15)) * lda + (k0 + ((lane >> 4) << 4));
  v16h a;
#pragma unroll
  for (int e = 0; e < 16; ++e) a[e] = p[e];
  return a;
}

// B fragment (32x16). Element (k,j) at Bm + k*sk + j*sj (runtime strides; used
// where strides are literal constants so they fold after inlining).
__device__ __forceinline__ v16h load_b_frag(const _Float16* __restrict__ Bm, long sk, long sj,
                                            int k0, int j0) {
  int lane = threadIdx.x & 31;
  const _Float16* p = Bm + (long)(k0 + ((lane >> 4) << 4)) * sk + (long)(j0 + (lane & 15)) * sj;
  v16h b;
#pragma unroll
  for (int e = 0; e < 16; ++e) b[e] = p[(long)e * sk];
  return b;
}

// ---------------------------------------------------------------------------
// Templated single-wave-per-tile WMMA GEMM: C[M,N] = A[M,K] * B + bias.
// Compile-time strides let SK==1 fragments merge into b128 loads; compile-time
// output flags keep the epilogue branch-free.
// ---------------------------------------------------------------------------
template <int SK, int SJ, bool HAS_BIAS, bool OUT_F32, bool OUT_F16>
__global__ void __launch_bounds__(32)
k_gemm_t(const _Float16* __restrict__ A, long lda,
         const _Float16* __restrict__ Bm,
         const float* __restrict__ bias,
         float* __restrict__ Cf, _Float16* __restrict__ Ch,
         long ldc, int K, int tilesN) {
  int tile = blockIdx.x;
  int m0 = (tile / tilesN) * 16;
  int n0 = (tile % tilesN) * 16;
  int lane = threadIdx.x & 31;
  int jc = lane & 15;
  int khi = lane >> 4;
  v8f acc = {};
  for (int k0 = 0; k0 < K; k0 += 32) {
    v16h a = load_a_frag(A, lda, m0, k0);
    const _Float16* p = Bm + (long)(k0 + (khi << 4)) * SK + (long)(n0 + jc) * SJ;
    v16h b;
#pragma unroll
    for (int e = 0; e < 16; ++e) b[e] = p[(long)e * SK];
    acc = wmma32(a, b, acc);
  }
  int col = n0 + jc;
  long rowb = m0 + (khi << 3);
  float bv = HAS_BIAS ? bias[col] : 0.0f;
#pragma unroll
  for (int r = 0; r < 8; ++r) {
    float v = acc[r] + bv;
    if (OUT_F32) Cf[(rowb + r) * ldc + col] = v;
    if (OUT_F16) Ch[(rowb + r) * ldc + col] = (_Float16)v;
  }
}

// ---------------------------------------------------------------------------
// Small elementwise / setup kernels
// ---------------------------------------------------------------------------
__global__ void k_fill_f32(float* p, float v, long n) {
  long i = (long)blockIdx.x * blockDim.x + threadIdx.x;
  if (i < n) p[i] = v;
}
__global__ void k_fill_f16(_Float16* p, float v, long n) {
  long i = (long)blockIdx.x * blockDim.x + threadIdx.x;
  if (i < n) p[i] = (_Float16)v;
}
__global__ void k_f32_to_f16(const float* __restrict__ s, _Float16* __restrict__ d, long n) {
  long i = (long)blockIdx.x * blockDim.x + threadIdx.x;
  if (i < n) d[i] = (_Float16)s[i];
}
__global__ void k_add_f32(const float* __restrict__ a, const float* __restrict__ b,
                          float* __restrict__ c, long n) {
  long i = (long)blockIdx.x * blockDim.x + threadIdx.x;
  if (i < n) c[i] = a[i] + b[i];
}
__global__ void k_rsqrt_f32(const float* __restrict__ s, float* __restrict__ d, long n) {
  long i = (long)blockIdx.x * blockDim.x + threadIdx.x;
  if (i < n) d[i] = rsqrtf(s[i]);
}

// ---------------------------------------------------------------------------
// GCN: degree, messages, normalized scatter (atomics), relu->f16
// ---------------------------------------------------------------------------
__global__ void k_deg_count(const int* __restrict__ ei, float* __restrict__ deg) {
  long i = (long)blockIdx.x * blockDim.x + threadIdx.x;
  if (i >= (long)B_ * E_) return;
  int b = (int)(i / E_), e = (int)(i % E_);
  int tgt = ei[(long)b * 2 * E_ + E_ + e];
  atomicAdd(&deg[b * N_ + tgt], 1.0f);
}

__global__ void k_gcn_mm1(const float* __restrict__ x, const float* __restrict__ W1,
                          float* __restrict__ m1) {
  long i = (long)blockIdx.x * blockDim.x + threadIdx.x;
  if (i >= (long)BN_ * H_) return;
  int h = (int)(i & (H_ - 1));
  long bn = i >> 8;
  const float* xp = x + bn * 4;
  m1[i] = xp[0] * W1[h] + xp[1] * W1[H_ + h] + xp[2] * W1[2 * H_ + h] + xp[3] * W1[3 * H_ + h];
}

// acc = dinv^2 * m  (self-loop) + bias
__global__ void k_scatter_init(const float* __restrict__ m, const float* __restrict__ dinv,
                               const float* __restrict__ bias, float* __restrict__ acc) {
  long i = (long)blockIdx.x * blockDim.x + threadIdx.x;
  if (i >= (long)BN_ * H_) return;
  int h = (int)(i & (H_ - 1));
  long bn = i >> 8;
  float dv = dinv[bn];
  acc[i] = dv * dv * m[i] + bias[h];
}

__global__ void k_edge_scatter(const float* __restrict__ m, const int* __restrict__ ei,
                               const float* __restrict__ dinv, float* __restrict__ acc) {
  long i = (long)blockIdx.x * blockDim.x + threadIdx.x;
  if (i >= (long)B_ * E_ * 64) return;
  int h4 = (int)(i & 63);
  long be = i >> 6;
  int e = (int)(be % E_), b = (int)(be / E_);
  const int* eb = ei + (long)b * 2 * E_;
  int s = eb[e], t = eb[E_ + e];
  float w = dinv[b * N_ + s] * dinv[b * N_ + t];
  const float* mp = m + ((long)b * N_ + s) * H_ + h4 * 4;
  float* ap = acc + ((long)b * N_ + t) * H_ + h4 * 4;
#pragma unroll
  for (int c = 0; c < 4; ++c) atomicAdd(ap + c, w * mp[c]);
}

__global__ void k_relu_f16(const float* __restrict__ s, _Float16* __restrict__ d, long n) {
  long i = (long)blockIdx.x * blockDim.x + threadIdx.x;
  if (i < n) d[i] = (_Float16)fmaxf(s[i], 0.0f);
}

// ---------------------------------------------------------------------------
// Flash-style attention: one wave per (batch, head, 16-query tile).
// K and V slices (224x64) are staged once into LDS with coalesced row chunks:
// K row-major (score B-frags = contiguous LDS), V transposed (PV B-frags =
// contiguous LDS). qkv rows: [B*N (+pad)], 768 cols (q | k | v).
// ---------------------------------------------------------------------------
__global__ void __launch_bounds__(32)
k_attn(const _Float16* __restrict__ qkv, _Float16* __restrict__ attn_o) {
  __shared__ _Float16 kbuf[224][64];    // K rows, row-major (28 KB)
  __shared__ _Float16 vbufT[64][224];   // V transposed      (28 KB)
  __shared__ float    sbuf[16][208];    // scores / exp      (13.3 KB)
  __shared__ _Float16 pbuf[16][224];    // softmax probs, K-padded (7 KB)
  int blk = blockIdx.x;
  int qt = blk % 13;
  int head = (blk / 13) & 3;
  int b = blk / 52;
  int lane = threadIdx.x;
  int jc = lane & 15;
  int khi = lane >> 4;

  // stage K and V (rows 0..223; pad rows exist past BN_, inert under P=0)
  for (int i = lane; i < 224 * 8; i += 32) {
    int r = i >> 3, c8 = (i & 7) << 3;
    const _Float16* pk = qkv + ((long)b * N_ + r) * 768 + H_ + head * 64 + c8;
    const _Float16* pv = pk + H_;
#pragma unroll
    for (int c = 0; c < 8; ++c) { kbuf[r][c8 + c] = pk[c]; vbufT[c8 + c][r] = pv[c]; }
  }

  // Q fragments (16 x 64)
  v16h aq0, aq1;
  {
    int qrow = qt * 16 + jc;
    if (qrow > N_ - 1) qrow = N_ - 1;
    const _Float16* p = qkv + ((long)b * N_ + qrow) * 768 + head * 64 + (khi << 4);
#pragma unroll
    for (int e = 0; e < 16; ++e) aq0[e] = p[e];
    p += 32;
#pragma unroll
    for (int e = 0; e < 16; ++e) aq1[e] = p[e];
  }
  __syncthreads();

  // S = Q K^T / sqrt(64), mask keys >= N
  for (int kt = 0; kt < 13; ++kt) {
    int krow = kt * 16 + jc;
    const _Float16* p = &kbuf[krow][khi << 4];
    v16h bk0, bk1;
#pragma unroll
    for (int e = 0; e < 16; ++e) { bk0[e] = p[e]; bk1[e] = p[e + 32]; }
    v8f s = {};
    s = wmma32(aq0, bk0, s);
    s = wmma32(aq1, bk1, s);
    int rb = khi << 3;
    bool oob = (krow >= N_);
#pragma unroll
    for (int r = 0; r < 8; ++r)
      sbuf[rb + r][kt * 16 + jc] = oob ? -1e30f : s[r] * 0.125f;
  }
  __syncthreads();

  // row softmax (lanes 0..15, one row each)
  if (lane < 16) {
    float mx = -1e30f;
    for (int c = 0; c < 208; ++c) mx = fmaxf(mx, sbuf[lane][c]);
    float sum = 0.0f;
    for (int c = 0; c < 208; ++c) { float e = __expf(sbuf[lane][c] - mx); sum += e; sbuf[lane][c] = e; }
    float inv = 1.0f / sum;
    for (int c = 0; c < 208; ++c) pbuf[lane][c] = (_Float16)(sbuf[lane][c] * inv);
    for (int c = 208; c < 224; ++c) pbuf[lane][c] = (_Float16)0.0f;
  }
  __syncthreads();

  // O = P V   (16 x 64, K = 224; padded K columns carry P = 0)
  v8f vz = {};
  v8f oacc[4];
#pragma unroll
  for (int nt = 0; nt < 4; ++nt) oacc[nt] = vz;
  for (int k0 = 0; k0 < 224; k0 += 32) {
    v16h a = load_a_frag(&pbuf[0][0], 224, 0, k0);
    int kb = k0 + (khi << 4);
#pragma unroll
    for (int nt = 0; nt < 4; ++nt) {
      const _Float16* p = &vbufT[nt * 16 + jc][kb];
      v16h bv;
#pragma unroll
      for (int e = 0; e < 16; ++e) bv[e] = p[e];
      oacc[nt] = wmma32(a, bv, oacc[nt]);
    }
  }
  int rb = khi << 3;
#pragma unroll
  for (int nt = 0; nt < 4; ++nt)
#pragma unroll
    for (int r = 0; r < 8; ++r) {
      int row = qt * 16 + rb + r;
      if (row < N_)
        attn_o[((long)b * N_ + row) * H_ + head * 64 + nt * 16 + jc] = (_Float16)oacc[nt][r];
    }
}

// mean over nodes (projection commutes with the mean)
__global__ void k_obar(const _Float16* __restrict__ attn_o, _Float16* __restrict__ obar) {
  int b = blockIdx.x, h = threadIdx.x;
  float s = 0.0f;
  for (int n = 0; n < N_; ++n) s += (float)attn_o[((long)b * N_ + n) * H_ + h];
  obar[b * H_ + h] = (_Float16)(s * (1.0f / (float)N_));
}

// ---------------------------------------------------------------------------
// 2-layer LSTM recurrence: 1 workgroup = (branch, 16 batch rows), all 30 steps.
// h in double-buffered LDS f16, c in VGPRs, weights f16 streamed from L2.
// Wave w owns hidden columns [w*64, w*64+64), incl. its i/f/g/o gate tiles.
// ---------------------------------------------------------------------------
__global__ void __launch_bounds__(128)
k_lstm(const float* __restrict__ x_feats, const float* __restrict__ Wih0f32,
       const _Float16* __restrict__ Wih1, const _Float16* __restrict__ Whh,
       const float* __restrict__ zc0, const float* __restrict__ bsum,
       float* __restrict__ last_h) {
  __shared__ _Float16 h0buf[2][16][H_];
  __shared__ _Float16 h1buf[2][16][H_];
  int kbr = blockIdx.x >> 4;
  int btile = blockIdx.x & 15;
  int wave = threadIdx.x >> 5;
  int lane = threadIdx.x & 31;
  int jc = lane & 15;
  int rb = (lane >> 4) << 3;

  for (int i = threadIdx.x; i < 2 * 16 * H_; i += blockDim.x) {
    (&h0buf[0][0][0])[i] = (_Float16)0.0f;
    (&h1buf[0][0][0])[i] = (_Float16)0.0f;
  }
  __syncthreads();

  v8f vz = {};
  v8f c0[4], c1[4];
#pragma unroll
  for (int i = 0; i < 4; ++i) { c0[i] = vz; c1[i] = vz; }

  const _Float16* whh0 = Whh + (long)(kbr * 2 + 0) * G4_ * H_;
  const _Float16* whh1 = Whh + (long)(kbr * 2 + 1) * G4_ * H_;
  const _Float16* wih1 = Wih1 + (long)kbr * G4_ * H_;
  const float* bs1 = bsum + (long)(kbr * 2 + 1) * G4_;
  const float* zc = zc0 + (long)kbr * B_ * G4_;
  int cur = 0;

  for (int t = 0; t < T_; ++t) {
    int nxt = cur ^ 1;
    // ---- layer 0: z = zc0(b,:) + x_t-part + h0 @ Whh0^T ----
#pragma unroll
    for (int ct = 0; ct < 4; ++ct) {
      int cbase = wave * 64 + ct * 16;
      v8f zi = vz, zf = vz, zg = vz, zo = vz;
      for (int k0 = 0; k0 < H_; k0 += 32) {
        v16h a = load_a_frag(&h0buf[cur][0][0], H_, 0, k0);
        zi = wmma32(a, load_b_frag(whh0, 1, H_, k0, 0 * H_ + cbase), zi);
        zf = wmma32(a, load_b_frag(whh0, 1, H_, k0, 1 * H_ + cbase), zf);
        zg = wmma32(a, load_b_frag(whh0, 1, H_, k0, 2 * H_ + cbase), zg);
        zo = wmma32(a, load_b_frag(whh0, 1, H_, k0, 3 * H_ + cbase), zo);
      }
      int col = cbase + jc;
      float w0g[4], w1g[4];
#pragma unroll
      for (int g = 0; g < 4; ++g) {
        long wb = ((long)kbr * G4_ + (g * H_ + col)) * 258;
        w0g[g] = Wih0f32[wb];
        w1g[g] = Wih0f32[wb + 1];
      }
#pragma unroll
      for (int r = 0; r < 8; ++r) {
        int brow = btile * 16 + rb + r;
        const float* xf = x_feats + (((long)kbr * B_ + brow) * T_ + t) * 2;
        float xf0 = xf[0], xf1 = xf[1];
        long zb = (long)brow * G4_ + col;
        float vi = zi[r] + zc[zb + 0 * H_] + xf0 * w0g[0] + xf1 * w1g[0];
        float vf = zf[r] + zc[zb + 1 * H_] + xf0 * w0g[1] + xf1 * w1g[1];
        float vg = zg[r] + zc[zb + 2 * H_] + xf0 * w0g[2] + xf1 * w1g[2];
        float vo = zo[r] + zc[zb + 3 * H_] + xf0 * w0g[3] + xf1 * w1g[3];
        float cn = sigmoidf_(vf) * c0[ct][r] + sigmoidf_(vi) * tanhf(vg);
        c0[ct][r] = cn;
        h0buf[nxt][rb + r][col] = (_Float16)(sigmoidf_(vo) * tanhf(cn));
      }
    }
    __syncthreads();
    // ---- layer 1: z = bias1 + h0_new @ Wih1^T + h1 @ Whh1^T ----
#pragma unroll
    for (int ct = 0; ct < 4; ++ct) {
      int cbase = wave * 64 + ct * 16;
      v8f zi = vz, zf = vz, zg = vz, zo = vz;
      for (int k0 = 0; k0 < H_; k0 += 32) {
        v16h a = load_a_frag(&h0buf[nxt][0][0], H_, 0, k0);
        zi = wmma32(a, load_b_frag(wih1, 1, H_, k0, 0 * H_ + cbase), zi);
        zf = wmma32(a, load_b_frag(wih1, 1, H_, k0, 1 * H_ + cbase), zf);
        zg = wmma32(a, load_b_frag(wih1, 1, H_, k0, 2 * H_ + cbase), zg);
        zo = wmma32(a, load_b_frag(wih1, 1, H_, k0, 3 * H_ + cbase), zo);
      }
      for (int k0 = 0; k0 < H_; k0 += 32) {
        v16h a = load_a_frag(&h1buf[cur][0][0], H_, 0, k0);
        zi = wmma32(a, load_b_frag(whh1, 1, H_, k0, 0 * H_ + cbase), zi);
        zf = wmma32(a, load_b_frag(whh1, 1, H_, k0, 1 * H_ + cbase), zf);
        zg = wmma32(a, load_b_frag(whh1, 1, H_, k0, 2 * H_ + cbase), zg);
        zo = wmma32(a, load_b_frag(whh1, 1, H_, k0, 3 * H_ + cbase), zo);
      }
      int col = cbase + jc;
      float b0 = bs1[0 * H_ + col], b1 = bs1[1 * H_ + col];
      float b2 = bs1[2 * H_ + col], b3 = bs1[3 * H_ + col];
#pragma unroll
      for (int r = 0; r < 8; ++r) {
        float vi = zi[r] + b0, vf = zf[r] + b1, vg = zg[r] + b2, vo = zo[r] + b3;
        float cn = sigmoidf_(vf) * c1[ct][r] + sigmoidf_(vi) * tanhf(vg);
        c1[ct][r] = cn;
        h1buf[nxt][rb + r][col] = (_Float16)(sigmoidf_(vo) * tanhf(cn));
      }
    }
    __syncthreads();
    cur = nxt;
  }

  for (int i = threadIdx.x; i < 16 * H_; i += blockDim.x) {
    int m = i >> 8, c = i & (H_ - 1);
    last_h[((long)kbr * B_ + btile * 16 + m) * H_ + c] = (float)h1buf[cur][m][c];
  }
}

// ---------------------------------------------------------------------------
// Output heads: one thread per (b, l, component); K=256 dot products (VALU)
// ---------------------------------------------------------------------------
__global__ void k_heads(const float* __restrict__ last_h,
                        const float* __restrict__ fcW_pv, const float* __restrict__ fcb_pv,
                        const float* __restrict__ fcW_s,  const float* __restrict__ fcb_s,
                        float* __restrict__ out) {
  int i = blockIdx.x * blockDim.x + threadIdx.x;
  if (i >= B_ * L_ * 7) return;
  int comp = i % 7;
  int l = (i / 7) % L_;
  int b = i / (7 * L_);
  const float* h;
  const float* w;
  float bias;
  bool sig = false;
  if (comp < 4) {
    int k = comp >> 1;                 // 0: position, 1: velocity
    int o = 2 * l + (comp & 1);
    h = last_h + (long)k * B_ * H_ + (long)b * H_;
    w = fcW_pv + (long)k * 2 * L_ * H_ + (long)o * H_;
    bias = fcb_pv[k * 2 * L_ + o];
  } else {
    int k = comp - 4;                  // 0: steering, 1: object, 2: traffic
    h = last_h + (long)(2 + k) * B_ * H_ + (long)b * H_;
    w = fcW_s + (long)k * L_ * H_ + (long)l * H_;
    bias = fcb_s[k * L_ + l];
    sig = (k >= 1);
  }
  float s = bias;
  for (int kk = 0; kk < H_; ++kk) s += h[kk] * w[kk];
  if (sig) s = sigmoidf_(s);
  out[i] = s;
}

// ---------------------------------------------------------------------------
// Host orchestration
// ---------------------------------------------------------------------------
extern "C" void kernel_launch(void* const* d_in, const int* in_sizes, int n_in,
                              void* d_out, int out_size, void* d_ws, size_t ws_size,
                              hipStream_t stream) {
  (void)in_sizes; (void)n_in; (void)out_size; (void)ws_size;
  const float* x_feats    = (const float*)d_in[0];
  const float* node_f     = (const float*)d_in[1];
  const int*   edge_idx   = (const int*)d_in[2];
  const float* gcn_W1     = (const float*)d_in[3];
  const float* gcn_b1     = (const float*)d_in[4];
  const float* gcn_W2     = (const float*)d_in[5];
  const float* gcn_b2     = (const float*)d_in[6];
  const float* attn_in_w  = (const float*)d_in[7];
  const float* attn_in_b  = (const float*)d_in[8];
  const float* attn_out_w = (const float*)d_in[9];
  const float* attn_out_b = (const float*)d_in[10];
  const float* Wih0       = (const float*)d_in[11];
  const float* Wih1       = (const float*)d_in[12];
  const float* Whh        = (const float*)d_in[13];
  const float* bih        = (const float*)d_in[14];
  const float* bhh        = (const float*)d_in[15];
  const float* fcW_pv     = (const float*)d_in[16];
  const float* fcb_pv     = (const float*)d_in[17];
  const float* fcW_s      = (const float*)d_in[18];
  const float* fcb_s      = (const float*)d_in[19];
  float* out = (float*)d_out;

  char* ws = (char*)d_ws;
  size_t off = 0;
  auto alloc = [&](size_t bytes) -> char* {
    char* p = ws + off;
    off += (bytes + 255) & ~(size_t)255;
    return p;
  };
  float*    deg     = (float*)alloc((size_t)BN_ * 4);
  float*    dinv    = (float*)alloc((size_t)BN_ * 4);
  _Float16* h1f16   = (_Float16*)alloc((size_t)BN_ * H_ * 2);   // later reused as attn_o
  _Float16* gf16    = (_Float16*)alloc((size_t)BN_ * H_ * 2);
  float*    mbuf    = (float*)alloc((size_t)BN_ * H_ * 4);      // messages; qkv aliases here
  float*    accbuf  = (float*)alloc((size_t)BN_ * H_ * 4);      // scatter accum; qkv tail
  _Float16* w2h     = (_Float16*)alloc((size_t)H_ * H_ * 2);
  _Float16* aiwh    = (_Float16*)alloc((size_t)768 * H_ * 2);
  _Float16* aowh    = (_Float16*)alloc((size_t)H_ * H_ * 2);
  _Float16* wih0h   = (_Float16*)alloc((size_t)5 * G4_ * 258 * 2);
  _Float16* wih1h   = (_Float16*)alloc((size_t)5 * G4_ * H_ * 2);
  _Float16* whhh    = (_Float16*)alloc((size_t)10 * G4_ * H_ * 2);
  float*    bsum    = (float*)alloc((size_t)10 * G4_ * 4);
  _Float16* obar    = (_Float16*)alloc((size_t)B_ * H_ * 2);
  _Float16* gfeat16 = (_Float16*)alloc((size_t)B_ * H_ * 2);
  float*    zc0     = (float*)alloc((size_t)5 * B_ * G4_ * 4);
  float*    lasth   = (float*)alloc((size_t)5 * B_ * H_ * 4);
  // aliases (regions dead by the time these are written):
  _Float16* qkv    = (_Float16*)mbuf;   // (BN_+256) x 768 f16 fits in mbuf+accbuf (105MB)
  _Float16* attn_o = h1f16;             // h1f16 dead after GCN layer-2 GEMM
  (void)accbuf;

  auto cdiv = [](long a, long b) -> int { return (int)((a + b - 1) / b); };

  // weight conversions + bias sums
  k_f32_to_f16<<<cdiv((long)H_ * H_, 256), 256, 0, stream>>>(gcn_W2, w2h, (long)H_ * H_);
  k_f32_to_f16<<<cdiv(768L * H_, 256), 256, 0, stream>>>(attn_in_w, aiwh, 768L * H_);
  k_f32_to_f16<<<cdiv((long)H_ * H_, 256), 256, 0, stream>>>(attn_out_w, aowh, (long)H_ * H_);
  k_f32_to_f16<<<cdiv(5L * G4_ * 258, 256), 256, 0, stream>>>(Wih0, wih0h, 5L * G4_ * 258);
  k_f32_to_f16<<<cdiv(5L * G4_ * H_, 256), 256, 0, stream>>>(Wih1, wih1h, 5L * G4_ * H_);
  k_f32_to_f16<<<cdiv(10L * G4_ * H_, 256), 256, 0, stream>>>(Whh, whhh, 10L * G4_ * H_);
  k_add_f32<<<cdiv(10L * G4_, 256), 256, 0, stream>>>(bih, bhh, bsum, 10L * G4_);

  // degrees / symmetric norm
  k_fill_f32<<<cdiv(BN_, 256), 256, 0, stream>>>(deg, 1.0f, BN_);
  k_deg_count<<<cdiv((long)B_ * E_, 256), 256, 0, stream>>>(edge_idx, deg);
  k_rsqrt_f32<<<cdiv(BN_, 256), 256, 0, stream>>>(deg, dinv, BN_);

  // GCN layer 1 (K=4: VALU) + scatter + relu
  k_gcn_mm1<<<cdiv((long)BN_ * H_, 256), 256, 0, stream>>>(node_f, gcn_W1, mbuf);
  k_scatter_init<<<cdiv((long)BN_ * H_, 256), 256, 0, stream>>>(mbuf, dinv, gcn_b1, accbuf);
  k_edge_scatter<<<cdiv((long)B_ * E_ * 64, 256), 256, 0, stream>>>(mbuf, edge_idx, dinv, accbuf);
  k_relu_f16<<<cdiv((long)BN_ * H_, 256), 256, 0, stream>>>(accbuf, h1f16, (long)BN_ * H_);

  // GCN layer 2: m2 = h1 @ W2  (B[k,j] = W2[k*H+j] -> SK=H, SJ=1)
  k_gemm_t<H_, 1, false, true, false><<<(BN_ / 16) * (H_ / 16), 32, 0, stream>>>(
      h1f16, H_, w2h, nullptr, mbuf, nullptr, H_, H_, H_ / 16);
  k_scatter_init<<<cdiv((long)BN_ * H_, 256), 256, 0, stream>>>(mbuf, dinv, gcn_b2, accbuf);
  k_edge_scatter<<<cdiv((long)B_ * E_ * 64, 256), 256, 0, stream>>>(mbuf, edge_idx, dinv, accbuf);
  k_relu_f16<<<cdiv((long)BN_ * H_, 256), 256, 0, stream>>>(accbuf, gf16, (long)BN_ * H_);

  // QKV projection: qkv = g @ attn_in_w^T + b  (SK=1, SJ=H)
  k_fill_f16<<<cdiv(256L * 768, 256), 256, 0, stream>>>(qkv + (size_t)BN_ * 768, 0.0f, 256L * 768);
  k_gemm_t<1, H_, true, false, true><<<(BN_ / 16) * (768 / 16), 32, 0, stream>>>(
      gf16, H_, aiwh, attn_in_b, nullptr, qkv, 768, H_, 768 / 16);

  // attention (flash-style, per (b, head, q-tile))
  k_attn<<<B_ * 4 * 13, 32, 0, stream>>>(qkv, attn_o);

  // mean over nodes, then tiny output projection -> gfeat (f16)
  k_obar<<<B_, H_, 0, stream>>>(attn_o, obar);
  k_gemm_t<1, H_, true, false, true><<<(B_ / 16) * (H_ / 16), 32, 0, stream>>>(
      obar, H_, aowh, attn_out_b, nullptr, gfeat16, H_, H_, H_ / 16);

  // zc0[k] = gfeat @ Wih0[k,:,2:]^T + (bih0+bhh0)   (time-invariant layer-0 input)
  for (int k = 0; k < 5; ++k)
    k_gemm_t<1, 258, true, true, false><<<(B_ / 16) * (G4_ / 16), 32, 0, stream>>>(
        gfeat16, H_, wih0h + (size_t)k * G4_ * 258 + 2,
        bsum + (size_t)(k * 2) * G4_, zc0 + (size_t)k * B_ * G4_, nullptr, G4_, H_, G4_ / 16);

  // 5-branch 2-layer LSTM recurrence (all 30 steps inside one kernel)
  k_lstm<<<5 * (B_ / 16), 128, 0, stream>>>(x_feats, Wih0, wih1h, whhh, zc0, bsum, lasth);

  // heads -> [B, L, 7]
  k_heads<<<cdiv((long)B_ * L_ * 7, 256), 256, 0, stream>>>(lasth, fcW_pv, fcb_pv, fcW_s, fcb_s, out);
}
